// AttentionBlock_19250043420853
// MI455X (gfx1250) — matrix operand order repaired and checked by reading, not compiled, not run on hardware
//
#include <hip/hip_runtime.h>
#include <hip/hip_bf16.h>
#include <math.h>

// ---------------------------------------------------------------------------
// CDNA5 (gfx1250, wave32) graph-transformer block.
// All GEMMs on v_wmma_f32_16x16x32_bf16 (bf16 in, f32 accumulate).
// GEMM-to-GEMM intermediates are stored bf16 to halve HBM traffic.
// ---------------------------------------------------------------------------

typedef __bf16 bf16_t;
typedef bf16_t bf16x16 __attribute__((ext_vector_type(16)));
typedef float  f32x8   __attribute__((ext_vector_type(8)));

union FragAB { uint4 q[2]; bf16x16 v; unsigned short s[16]; };

__device__ __forceinline__ unsigned short f2bf(float f) {
  unsigned int u = __float_as_uint(f);
  u += 0x7FFFu + ((u >> 16) & 1u);       // round-to-nearest-even
  return (unsigned short)(u >> 16);
}
__device__ __forceinline__ float bf2f(unsigned short h) {
  return __uint_as_float((unsigned int)h << 16);
}
__device__ __forceinline__ float gelu_f(float x) {
  return 0.5f * x * (1.0f + erff(x * 0.7071067811865475f));
}
__device__ __forceinline__ void atomicMaxF(float* addr, float val) {
  unsigned int* ua = (unsigned int*)addr;
  unsigned int old = *(volatile unsigned int*)ua;
  while (__uint_as_float(old) < val) {
    unsigned int assumed = old;
    old = atomicCAS(ua, assumed, __float_as_uint(val));
    if (old == assumed) break;
  }
}

// ---------------------------------------------------------------------------
// Generic WMMA GEMM: C[M,N] = act(A[M,K] @ B[K,N] + bias (+ resid))
//   Bt = B transposed to [N][K] bf16 (fragment loads contiguous).
//   ECAT : A row e is the fused edge feature [edge_attr | nx-ny | nx+ny]
//   ABF16: A is already bf16 (stage-in = raw 32B copy, no conversion)
//   OBF16: store C as bf16
// Block = 256 threads (8 waves), tile = 128x64, K step = 32.
// ---------------------------------------------------------------------------
#define BM  128
#define BN  64
#define LDT 40   // LDS row stride in halves (32 + 8 pad, 80B keeps 16B align)

template<int ACT, int ECAT, int ABF16, int OBF16>
__global__ __launch_bounds__(256)
void gemm_kernel(const void* __restrict__ Avoid, const unsigned short* __restrict__ Bt,
                 const float* __restrict__ bias, const float* __restrict__ resid,
                 void* __restrict__ Cvoid, int M, int N, int K,
                 const float* __restrict__ eattr, const float* __restrict__ nodef,
                 const int* __restrict__ erow, const int* __restrict__ ecol)
{
  __shared__ __attribute__((aligned(16))) unsigned short As[BM * LDT];
  __shared__ __attribute__((aligned(16))) unsigned short Bs[BN * LDT];

  const float*          Af = (const float*)Avoid;
  const unsigned short* Ab = (const unsigned short*)Avoid;
  float*                Cf = (float*)Cvoid;
  unsigned short*       Cb = (unsigned short*)Cvoid;

  const int tid  = threadIdx.x;
  const int lane = tid & 31;
  const int wave = tid >> 5;
  const int wm   = wave & 3;    // wave row (4 x 32 = 128)
  const int wn   = wave >> 2;   // wave col (2 x 32 = 64)
  const int l16  = lane & 15;
  const int hi   = lane >> 4;   // lane half -> K phase of fragments

  const int bm0 = blockIdx.x * BM;
  const int bn0 = blockIdx.y * BN;

  f32x8 acc00 = {}, acc01 = {}, acc10 = {}, acc11 = {};

  const int ar = tid >> 1;      // A tile row (0..127)
  const int ah = tid & 1;       // 16-col half
  const int br = tid >> 2;      // Bt tile row (0..63)
  const int bq = tid & 3;       // 8-half chunk

  int jr = 0, jc = 0;
  if (ECAT) {
    int e = bm0 + ar;
    if (e < M) { jr = erow[e]; jc = ecol[e]; }
  }

  for (int kc = 0; kc < K; kc += 32) {
    // ---- stage A tile: BM x 32 -> bf16 LDS ----
    {
      union { unsigned short s[16]; uint4 q[2]; } tmp;
      const int gr = bm0 + ar;
      const int c0 = kc + ah * 16;
      if (ECAT) {
        // fused ecat = [edge_attr(64) | nx-ny(128) | nx+ny(128)], K = 320
        #pragma unroll
        for (int i = 0; i < 16; ++i) {
          int c = c0 + i;
          float val = 0.f;
          if (gr < M) {
            if (c < 64) {
              val = eattr[(size_t)gr * 64 + c];
            } else {
              int cc = (c - 64) & 127;
              float a = nodef[(size_t)jr * 128 + cc];
              float b = nodef[(size_t)jc * 128 + cc];
              val = (c < 192) ? (a - b) : (a + b);
            }
          }
          tmp.s[i] = f2bf(val);
        }
      } else if (ABF16) {
        if (gr < M) {
          const uint4* src = (const uint4*)(Ab + (size_t)gr * K + c0);
          tmp.q[0] = src[0]; tmp.q[1] = src[1];
        } else {
          tmp.q[0] = make_uint4(0,0,0,0); tmp.q[1] = make_uint4(0,0,0,0);
        }
      } else {
        if (gr < M) {
          const float4* src = (const float4*)(Af + (size_t)gr * K + c0);
          if (kc + 32 < K) __builtin_prefetch(Af + (size_t)gr * K + kc + 32, 0, 1);
          #pragma unroll
          for (int i = 0; i < 4; ++i) {
            float4 v = src[i];
            tmp.s[i*4+0] = f2bf(v.x); tmp.s[i*4+1] = f2bf(v.y);
            tmp.s[i*4+2] = f2bf(v.z); tmp.s[i*4+3] = f2bf(v.w);
          }
        } else {
          #pragma unroll
          for (int i = 0; i < 16; ++i) tmp.s[i] = 0;
        }
      }
      uint4* dst = (uint4*)&As[ar * LDT + ah * 16];
      dst[0] = tmp.q[0]; dst[1] = tmp.q[1];
    }
    // ---- stage Bt tile: BN x 32 bf16 ----
    {
      const uint4* src = (const uint4*)(Bt + (size_t)(bn0 + br) * K + kc + bq * 8);
      *(uint4*)&Bs[br * LDT + bq * 8] = *src;
    }
    __syncthreads();

    // ---- fragments per ISA 16-bit layouts; 2 x ds_load_b128 each ----
    FragAB a0, a1, b0, b1;
    {
      const unsigned short* ap0 = &As[(wm*32 + l16) * LDT + hi*8];
      a0.q[0] = *(const uint4*)(ap0);
      a0.q[1] = *(const uint4*)(ap0 + 16);
      const unsigned short* ap1 = &As[(wm*32 + 16 + l16) * LDT + hi*8];
      a1.q[0] = *(const uint4*)(ap1);
      a1.q[1] = *(const uint4*)(ap1 + 16);
      const unsigned short* bp0 = &Bs[(wn*32 + l16) * LDT + hi*8];
      b0.q[0] = *(const uint4*)(bp0);
      b0.q[1] = *(const uint4*)(bp0 + 16);
      const unsigned short* bp1 = &Bs[(wn*32 + 16 + l16) * LDT + hi*8];
      b1.q[0] = *(const uint4*)(bp1);
      b1.q[1] = *(const uint4*)(bp1 + 16);
    }
    acc00 = __builtin_amdgcn_wmma_f32_16x16x32_bf16(false, a0.v, false, b0.v, (short)0, acc00, false, false);
    acc01 = __builtin_amdgcn_wmma_f32_16x16x32_bf16(false, a0.v, false, b1.v, (short)0, acc01, false, false);
    acc10 = __builtin_amdgcn_wmma_f32_16x16x32_bf16(false, a1.v, false, b0.v, (short)0, acc10, false, false);
    acc11 = __builtin_amdgcn_wmma_f32_16x16x32_bf16(false, a1.v, false, b1.v, (short)0, acc11, false, false);
    __syncthreads();
  }

  // ---- epilogue: bias / residual / activation / store (f32 or bf16) ----
  const float bias_n0 = bias ? bias[bn0 + wn*32 + l16]      : 0.f;
  const float bias_n1 = bias ? bias[bn0 + wn*32 + 16 + l16] : 0.f;
  #pragma unroll
  for (int i = 0; i < 2; ++i) {
    f32x8 aj0 = i ? acc10 : acc00;
    f32x8 aj1 = i ? acc11 : acc01;
    #pragma unroll
    for (int r = 0; r < 8; ++r) {
      int m = bm0 + wm*32 + i*16 + hi*8 + r;
      if (m < M) {
        int n0 = bn0 + wn*32 + l16;
        float v0 = aj0[r] + bias_n0;
        if (resid) v0 += resid[(size_t)m * N + n0];
        if (ACT == 1) v0 = gelu_f(v0);
        int n1 = n0 + 16;
        float v1 = aj1[r] + bias_n1;
        if (resid) v1 += resid[(size_t)m * N + n1];
        if (ACT == 1) v1 = gelu_f(v1);
        if (OBF16) {
          Cb[(size_t)m * N + n0] = f2bf(v0);
          Cb[(size_t)m * N + n1] = f2bf(v1);
        } else {
          Cf[(size_t)m * N + n0] = v0;
          Cf[(size_t)m * N + n1] = v1;
        }
      }
    }
  }
}

// ---------------------------------------------------------------------------
// Small helper kernels
// ---------------------------------------------------------------------------
__global__ void fill_kernel(float* p, float v, long n) {
  long t = (long)blockIdx.x * blockDim.x + threadIdx.x;
  if (t < n) p[t] = v;
}

// W [K,N] f32 row-major -> Bt [N,K] bf16
__global__ void wt_prep_kernel(const float* __restrict__ W, unsigned short* __restrict__ Bt,
                               int K, int N) {
  int t = blockIdx.x * blockDim.x + threadIdx.x;
  if (t >= K * N) return;
  int n = t / K, k = t % K;
  Bt[(size_t)n * K + k] = f2bf(W[(size_t)k * N + n]);
}

__global__ void concat4_kernel(const float* a, const float* b, const float* c,
                               const float* d, float* o) {
  int t = blockIdx.x * blockDim.x + threadIdx.x;
  if (t < 128)      o[t] = a[t];
  else if (t < 256) o[t] = b[t - 128];
  else if (t < 384) o[t] = c[t - 256];
  else if (t < 512) o[t] = d[t - 384];
}

// attn_out init = skip projection slice of fused qkvs
__global__ void attn_init_kernel(const float* __restrict__ qkvs, float* __restrict__ out, int Nn) {
  long t = (long)blockIdx.x * 256 + threadIdx.x;
  if (t >= (long)Nn * 128) return;
  int n = (int)(t >> 7), c = (int)(t & 127);
  out[t] = qkvs[(size_t)n * 512 + 384 + c];
}

// alpha[e,h] = dot(q[col], k[row]+e) / 4 ; segment max via atomic.  e is bf16.
__global__ void alpha_kernel(const float* __restrict__ qkvs, const unsigned short* __restrict__ ebuf,
                             const int* __restrict__ erow, const int* __restrict__ ecol,
                             float* __restrict__ alpha, float* __restrict__ amax, int E) {
  long t = (long)blockIdx.x * 256 + threadIdx.x;
  if (t >= (long)E * 8) return;
  int e = (int)(t >> 3), h = (int)(t & 7);
  int ri = erow[e], ci = ecol[e];
  const float4* qp = (const float4*)(qkvs + (size_t)ci * 512 + h * 16);
  const float4* kp = (const float4*)(qkvs + (size_t)ri * 512 + 128 + h * 16);
  const unsigned int* ep = (const unsigned int*)(ebuf + (size_t)e * 128 + h * 16);
  float acc = 0.f;
  #pragma unroll
  for (int i = 0; i < 4; ++i) {
    float4 q = qp[i], k = kp[i];
    unsigned int u0 = ep[i*2], u1 = ep[i*2+1];
    acc += q.x*(k.x + bf2f((unsigned short)(u0 & 0xffff)))
         + q.y*(k.y + bf2f((unsigned short)(u0 >> 16)))
         + q.z*(k.z + bf2f((unsigned short)(u1 & 0xffff)))
         + q.w*(k.w + bf2f((unsigned short)(u1 >> 16)));
  }
  acc *= 0.25f;   // 1/sqrt(16)
  alpha[t] = acc;
  atomicMaxF(&amax[(size_t)ci * 8 + h], acc);
}

__global__ void expsum_kernel(float* __restrict__ alpha, const float* __restrict__ amax,
                              const int* __restrict__ ecol, float* __restrict__ asum, int E) {
  long t = (long)blockIdx.x * 256 + threadIdx.x;
  if (t >= (long)E * 8) return;
  int e = (int)(t >> 3), h = (int)(t & 7);
  int ci = ecol[e];
  float ex = __expf(alpha[t] - amax[(size_t)ci * 8 + h]);
  alpha[t] = ex;
  atomicAdd(&asum[(size_t)ci * 8 + h], ex);
}

// scatter msg = (v[row]+e) * softmax weight into attn_out[col]
__global__ void msg_kernel(const float* __restrict__ qkvs, const unsigned short* __restrict__ ebuf,
                           const float* __restrict__ alpha, const float* __restrict__ asum,
                           const int* __restrict__ erow, const int* __restrict__ ecol,
                           float* __restrict__ out, int E) {
  long t = (long)blockIdx.x * 256 + threadIdx.x;
  if (t >= (long)E * 8) return;
  int e = (int)(t >> 3), h = (int)(t & 7);
  int ri = erow[e], ci = ecol[e];
  float w = alpha[t] / (asum[(size_t)ci * 8 + h] + 1e-16f);
  const float4* vp = (const float4*)(qkvs + (size_t)ri * 512 + 256 + h * 16);
  const unsigned int* ep = (const unsigned int*)(ebuf + (size_t)e * 128 + h * 16);
  float* op = out + (size_t)ci * 128 + h * 16;
  #pragma unroll
  for (int i = 0; i < 4; ++i) {
    float4 v = vp[i];
    unsigned int u0 = ep[i*2], u1 = ep[i*2+1];
    atomicAdd(op + i*4 + 0, (v.x + bf2f((unsigned short)(u0 & 0xffff))) * w);
    atomicAdd(op + i*4 + 1, (v.y + bf2f((unsigned short)(u0 >> 16)))    * w);
    atomicAdd(op + i*4 + 2, (v.z + bf2f((unsigned short)(u1 & 0xffff))) * w);
    atomicAdd(op + i*4 + 3, (v.w + bf2f((unsigned short)(u1 >> 16)))    * w);
  }
}

// per-column sum / sumsq (C in {64,128}), partial + atomic
__global__ __launch_bounds__(256)
void colstats_kernel(const float* __restrict__ X, float* __restrict__ sums,
                     int M, int C, int rowsPerBlock) {
  int tid = threadIdx.x;
  int c = tid % C;
  int rl = tid / C;
  int rstep = 256 / C;
  long base = (long)blockIdx.x * rowsPerBlock;
  float s = 0.f, s2 = 0.f;
  for (int r = rl; r < rowsPerBlock; r += rstep) {
    long row = base + r;
    if (row < M) { float x = X[row * C + c]; s += x; s2 += x * x; }
  }
  __shared__ float sh[512];
  sh[tid] = s; sh[256 + tid] = s2;
  __syncthreads();
  if (rl == 0) {
    for (int i = 1; i < rstep; ++i) { s += sh[i * C + c]; s2 += sh[256 + i * C + c]; }
    atomicAdd(&sums[c], s);
    atomicAdd(&sums[C + c], s2);
  }
}

__global__ void bn_apply_kernel(const float* __restrict__ X, const float* __restrict__ sums,
                                const float* __restrict__ g, const float* __restrict__ b,
                                float* __restrict__ out_raw, float* __restrict__ out_act,
                                long M, int C) {
  long t = (long)blockIdx.x * 256 + threadIdx.x;
  if (t >= M * C) return;
  int c = (int)(t % C);
  float mean = sums[c] / (float)M;
  float var  = sums[C + c] / (float)M - mean * mean;
  float y = (X[t] - mean) * rsqrtf(var + 1e-5f) * g[c] + b[c];
  if (out_raw) out_raw[t] = y;
  if (out_act) out_act[t] = gelu_f(y);
}

// ---------------------------------------------------------------------------
extern "C" void kernel_launch(void* const* d_in, const int* in_sizes, int n_in,
                              void* d_out, int out_size, void* d_ws, size_t ws_size,
                              hipStream_t stream) {
  const float* node_attr = (const float*)d_in[0];
  const float* edge_attr = (const float*)d_in[1];
  const int*   eidx      = (const int*)d_in[2];
  // d_in[3] = batch (unused)
  const float* wq    = (const float*)d_in[4];
  const float* bq    = (const float*)d_in[5];
  const float* wk    = (const float*)d_in[6];
  const float* bk    = (const float*)d_in[7];
  const float* wv    = (const float*)d_in[8];
  const float* bv    = (const float*)d_in[9];
  const float* we    = (const float*)d_in[10];
  const float* wskip = (const float*)d_in[11];
  const float* bskip = (const float*)d_in[12];
  const float* w_ff1 = (const float*)d_in[13];
  const float* b_ff1 = (const float*)d_in[14];
  const float* w_ff2 = (const float*)d_in[15];
  const float* b_ff2 = (const float*)d_in[16];
  const float* w_e1  = (const float*)d_in[17];
  const float* b_e1  = (const float*)d_in[18];
  const float* w_e2  = (const float*)d_in[19];
  const float* b_e2  = (const float*)d_in[20];
  const float* g1    = (const float*)d_in[21];
  const float* bt1   = (const float*)d_in[22];
  const float* g2    = (const float*)d_in[23];
  const float* bt2   = (const float*)d_in[24];
  const float* geW   = (const float*)d_in[25];
  const float* bte   = (const float*)d_in[26];

  const int Nn = in_sizes[0] / 128;   // 50000 nodes
  const int E  = in_sizes[1] / 64;    // 800000 edges
  const int* erow = eidx;             // source j
  const int* ecol = eidx + E;         // target i

  // workspace layout (~710 MB)
  float* ws = (float*)d_ws;
  size_t off = 0;
  auto alloc = [&](size_t nf) -> float* {
    float* p = ws + off; off += (nf + 255) & ~(size_t)255; return p;
  };
  float* qkvs        = alloc((size_t)Nn * 512);          // [q|k|v|skip] f32
  unsigned short* e_buf = (unsigned short*)alloc((size_t)E * 64);  // E x 128 bf16; reused
  float* eo2         = alloc((size_t)E * 64);            // f32 (BN stats)
  float* alpha       = alloc((size_t)E * 8);
  float* amax        = alloc((size_t)Nn * 8);
  float* asum        = alloc((size_t)Nn * 8);
  float* attn_out    = alloc((size_t)Nn * 128);
  float* node_out    = alloc((size_t)Nn * 128);
  unsigned short* ff_hidden = (unsigned short*)alloc((size_t)Nn * 512); // Nn x 1024 bf16
  float* ff_out      = alloc((size_t)Nn * 128);
  float* node_out2   = alloc((size_t)Nn * 128);
  float* bnsums      = alloc(256);
  float* bias_qkvs   = alloc(512);
  unsigned short* Btqkvs = (unsigned short*)alloc(181000);
  unsigned short* Btwe  = Btqkvs + 512 * 128;
  unsigned short* Btff1 = Btwe   + 128 * 64;
  unsigned short* Btff2 = Btff1  + 1024 * 128;
  unsigned short* Bte1  = Btff2  + 128 * 1024;
  unsigned short* Bte2  = Bte1   + 64 * 320;
  unsigned short* eo_hidden = e_buf;   // reuse: E x 64 bf16 (edge proj consumed)

  float* out_node = (float*)d_out;
  float* out_edge = out_node + (size_t)Nn * 128;

  const dim3 blk(256);
  const float NEG_INF = -__builtin_inff();
  const void* NP = nullptr;

  // --- weight prep: transpose + f32->bf16 ---
  wt_prep_kernel<<<(128*128+255)/256, blk, 0, stream>>>(wq,    Btqkvs + 0*128*128, 128, 128);
  wt_prep_kernel<<<(128*128+255)/256, blk, 0, stream>>>(wk,    Btqkvs + 1*128*128, 128, 128);
  wt_prep_kernel<<<(128*128+255)/256, blk, 0, stream>>>(wv,    Btqkvs + 2*128*128, 128, 128);
  wt_prep_kernel<<<(128*128+255)/256, blk, 0, stream>>>(wskip, Btqkvs + 3*128*128, 128, 128);
  wt_prep_kernel<<<(64*128+255)/256,   blk, 0, stream>>>(we,    Btwe,   64, 128);
  wt_prep_kernel<<<(128*1024+255)/256, blk, 0, stream>>>(w_ff1, Btff1, 128, 1024);
  wt_prep_kernel<<<(1024*128+255)/256, blk, 0, stream>>>(w_ff2, Btff2, 1024, 128);
  wt_prep_kernel<<<(320*64+255)/256,   blk, 0, stream>>>(w_e1,  Bte1,  320, 64);
  wt_prep_kernel<<<(64*64+255)/256,    blk, 0, stream>>>(w_e2,  Bte2,  64, 64);
  concat4_kernel<<<2, blk, 0, stream>>>(bq, bk, bv, bskip, bias_qkvs);

  // --- fused Q/K/V/skip projection: [Nn,128] @ [128,512] -> f32 ---
  gemm_kernel<0,0,0,0><<<dim3((Nn + 127)/128, 8), blk, 0, stream>>>(
      node_attr, Btqkvs, bias_qkvs, nullptr, qkvs, Nn, 512, 128,
      nullptr, nullptr, nullptr, nullptr);

  // --- edge projection: [E,64] @ [64,128] -> bf16 ---
  gemm_kernel<0,0,0,1><<<dim3(E/128, 2), blk, 0, stream>>>(
      edge_attr, Btwe, nullptr, nullptr, e_buf, E, 128, 64,
      nullptr, nullptr, nullptr, nullptr);

  // --- attention ---
  fill_kernel<<<((long)Nn*8 + 255)/256, blk, 0, stream>>>(amax, NEG_INF, (long)Nn*8);
  fill_kernel<<<((long)Nn*8 + 255)/256, blk, 0, stream>>>(asum, 0.f,     (long)Nn*8);
  attn_init_kernel<<<((long)Nn*128 + 255)/256, blk, 0, stream>>>(qkvs, attn_out, Nn);
  alpha_kernel <<<((long)E*8 + 255)/256, blk, 0, stream>>>(qkvs, e_buf, erow, ecol, alpha, amax, E);
  expsum_kernel<<<((long)E*8 + 255)/256, blk, 0, stream>>>(alpha, amax, ecol, asum, E);
  msg_kernel   <<<((long)E*8 + 255)/256, blk, 0, stream>>>(qkvs, e_buf, alpha, asum, erow, ecol, attn_out, E);

  // --- BN1 ---
  fill_kernel<<<1, blk, 0, stream>>>(bnsums, 0.f, 256);
  colstats_kernel<<<(Nn + 255)/256, blk, 0, stream>>>(attn_out, bnsums, Nn, 128, 256);
  bn_apply_kernel<<<((long)Nn*128 + 255)/256, blk, 0, stream>>>(
      attn_out, bnsums, g1, bt1, node_out, nullptr, Nn, 128);

  // --- FF MLP: gelu GEMM -> bf16 hidden -> bf16-A GEMM + residual ---
  gemm_kernel<1,0,0,1><<<dim3((Nn + 127)/128, 16), blk, 0, stream>>>(
      node_out, Btff1, b_ff1, nullptr, ff_hidden, Nn, 1024, 128,
      nullptr, nullptr, nullptr, nullptr);
  gemm_kernel<0,0,1,0><<<dim3((Nn + 127)/128, 2), blk, 0, stream>>>(
      ff_hidden, Btff2, b_ff2, node_out, ff_out, Nn, 128, 1024,
      nullptr, nullptr, nullptr, nullptr);

  // --- BN2: raw -> node_out2 (feeds edge MLP), gelu -> final node output ---
  fill_kernel<<<1, blk, 0, stream>>>(bnsums, 0.f, 256);
  colstats_kernel<<<(Nn + 255)/256, blk, 0, stream>>>(ff_out, bnsums, Nn, 128, 256);
  bn_apply_kernel<<<((long)Nn*128 + 255)/256, blk, 0, stream>>>(
      ff_out, bnsums, g2, bt2, node_out2, out_node, Nn, 128);

  // --- edge MLP: fused ecat gather GEMM (gelu, bf16 out) -> bf16-A GEMM ---
  gemm_kernel<1,1,0,1><<<dim3(E/128, 1), blk, 0, stream>>>(
      NP, Bte1, b_e1, nullptr, eo_hidden, E, 64, 320,
      edge_attr, node_out2, erow, ecol);
  gemm_kernel<0,0,1,0><<<dim3(E/128, 1), blk, 0, stream>>>(
      eo_hidden, Bte2, b_e2, nullptr, eo2, E, 64, 64,
      nullptr, nullptr, nullptr, nullptr);

  // --- edge BN + final gelu ---
  fill_kernel<<<1, blk, 0, stream>>>(bnsums, 0.f, 256);
  colstats_kernel<<<(E + 255)/256, blk, 0, stream>>>(eo2, bnsums, E, 64, 256);
  bn_apply_kernel<<<((long)E*64 + 255)/256, blk, 0, stream>>>(
      eo2, bnsums, geW, bte, nullptr, out_edge, E, 64);
}
